// GNNModel_13202729468198
// MI455X (gfx1250) — matrix operand order repaired and checked by reading, not compiled, not run on hardware
//
#include <hip/hip_runtime.h>

#define N_NODES 40000
#define N_EDGES 640000
#define HID 256

typedef __attribute__((ext_vector_type(16))) _Float16 v16h;
typedef __attribute__((ext_vector_type(8)))  _Float16 v8h;
typedef __attribute__((ext_vector_type(8)))  float    v8f;

// ---------------- zero fill ----------------
__global__ void zero_kernel(float* __restrict__ p, int n) {
  int i = blockIdx.x * blockDim.x + threadIdx.x;
  if (i < n) p[i] = 0.0f;
}

// ---------------- edge scatter-add (segment_sum) ----------------
// One thread handles one (edge, float4) chunk: consecutive threads cover one
// edge's row of x[src] -> coalesced gather, then 4 native f32 atomics.
__global__ void scatter_add_kernel(const float* __restrict__ x,
                                   const long long* __restrict__ src,
                                   const long long* __restrict__ dst,
                                   float* __restrict__ agg, int dim) {
  const int vpe = dim >> 2;  // float4s per edge
  long long idx = (long long)blockIdx.x * blockDim.x + threadIdx.x;
  long long total = (long long)N_EDGES * vpe;
  if (idx >= total) return;
  int e = (int)(idx / vpe);
  int v = (int)(idx % vpe);
  long long s = src[e];
  long long d = dst[e];
  const float4 xv = *(const float4*)(x + s * dim + (long long)v * 4);
  float* ap = agg + d * dim + (long long)v * 4;
  unsafeAtomicAdd(ap + 0, xv.x);
  unsafeAtomicAdd(ap + 1, xv.y);
  unsafeAtomicAdd(ap + 2, xv.z);
  unsafeAtomicAdd(ap + 3, xv.w);
}

// ---------------- h = (1+eps)*x + agg, cast to f16 ----------------
__global__ void combine_cast_kernel(const float* __restrict__ x,
                                    const float* __restrict__ agg,
                                    const float* __restrict__ eps,
                                    _Float16* __restrict__ h, int n) {
  int i = blockIdx.x * blockDim.x + threadIdx.x;
  if (i < n) {
    float e = 1.0f + eps[0];
    h[i] = (_Float16)(e * x[i] + agg[i]);
  }
}

// ---------------- W (K x 256, f32) -> Wt (256 x K, f16) ----------------
__global__ void transpose_cast_kernel(const float* __restrict__ W,
                                      _Float16* __restrict__ Wt, int K) {
  int i = blockIdx.x * blockDim.x + threadIdx.x;
  if (i < K * HID) {
    int k = i / HID;
    int o = i % HID;
    Wt[(long long)o * K + k] = (_Float16)W[i];
  }
}

// ---------------- WMMA GEMM: C = relu(A @ Wt^T + bias) ----------------
// A: M x K (f16 row-major), Bt: 256 x K (f16, row n = output column n),
// C: M x 256 (OutT). Each wave computes a 16x64 stripe: 4 accumulators,
// one A fragment reused across 4 WMMAs per K-chunk. 4 waves/block cover
// all 256 columns; grid.x covers M/16 row tiles.
// A-frag  (16x32 f16): lane = M row (lane&15), hi=lane>>4; halves 0-7 ->
//   K = kk + hi*8 + 0..7, halves 8-15 -> K = kk + 16 + hi*8 + 0..7.
// B-frag  (32x16 f16): lane = N col (lane&15); same K order as A-frag.
// C/D f32: lane col = lane&15; float i -> row hi*8 + i.
template <typename OutT>
__global__ void wmma_gemm_relu(const _Float16* __restrict__ A,
                               const _Float16* __restrict__ Bt,
                               const float* __restrict__ bias,
                               OutT* __restrict__ C, int K) {
  const int lane = threadIdx.x & 31;
  const int wave = threadIdx.x >> 5;
  const int r    = lane & 15;
  const int hi   = lane >> 4;
  const int m0   = blockIdx.x * 16;
  const int n0   = wave * 64;  // this wave's 64-column stripe

  const _Float16* arow = A + (long long)(m0 + r) * K;

  v8f acc[4];
#pragma unroll
  for (int j = 0; j < 4; ++j) {
    const float bv = bias[n0 + j * 16 + r];
#pragma unroll
    for (int i = 0; i < 8; ++i) acc[j][i] = bv;
  }

  const _Float16* brow[4];
#pragma unroll
  for (int j = 0; j < 4; ++j)
    brow[j] = Bt + (long long)(n0 + j * 16 + r) * K;

#pragma unroll 2
  for (int kk = 0; kk < K; kk += 32) {
    v16h a;
    ((v8h*)&a)[0] = *(const v8h*)(arow + kk + hi * 8);
    ((v8h*)&a)[1] = *(const v8h*)(arow + kk + 16 + hi * 8);
#pragma unroll
    for (int j = 0; j < 4; ++j) {
      v16h b;
      ((v8h*)&b)[0] = *(const v8h*)(brow[j] + kk + hi * 8);
      ((v8h*)&b)[1] = *(const v8h*)(brow[j] + kk + 16 + hi * 8);
      acc[j] = __builtin_amdgcn_wmma_f32_16x16x32_f16(
          /*neg_a=*/false, a, /*neg_b=*/false, b,
          /*c_mod=*/(short)0, acc[j], /*reuse_a=*/false, /*reuse_b=*/false);
    }
  }

#pragma unroll
  for (int j = 0; j < 4; ++j) {
#pragma unroll
    for (int i = 0; i < 8; ++i) {
      int row = m0 + hi * 8 + i;
      float v = acc[j][i];
      v = v > 0.0f ? v : 0.0f;
      C[(long long)row * HID + n0 + j * 16 + r] = (OutT)v;
    }
  }
}

// ---------------- host launch ----------------
static inline size_t align256(size_t x) { return (x + 255) & ~(size_t)255; }

extern "C" void kernel_launch(void* const* d_in, const int* in_sizes, int n_in,
                              void* d_out, int out_size, void* d_ws, size_t ws_size,
                              hipStream_t stream) {
  const float* x0 = (const float*)d_in[0];
  const long long* ei = (const long long*)d_in[1];
  const long long* src = ei;              // edge_index[0]
  const long long* dst = ei + N_EDGES;    // edge_index[1]

  const float* W1[3] = {(const float*)d_in[2],  (const float*)d_in[7],  (const float*)d_in[12]};
  const float* b1[3] = {(const float*)d_in[3],  (const float*)d_in[8],  (const float*)d_in[13]};
  const float* W2[3] = {(const float*)d_in[4],  (const float*)d_in[9],  (const float*)d_in[14]};
  const float* b2[3] = {(const float*)d_in[5],  (const float*)d_in[10], (const float*)d_in[15]};
  const float* eps[3] = {(const float*)d_in[6], (const float*)d_in[11], (const float*)d_in[16]};

  char* ws = (char*)d_ws;
  float* agg = (float*)ws;          ws += align256((size_t)N_NODES * HID * sizeof(float));
  _Float16* hf = (_Float16*)ws;     ws += align256((size_t)N_NODES * HID * sizeof(_Float16));
  _Float16* tf = (_Float16*)ws;     ws += align256((size_t)N_NODES * HID * sizeof(_Float16));
  _Float16* w1t = (_Float16*)ws;    ws += align256((size_t)HID * HID * sizeof(_Float16));
  _Float16* w2t = (_Float16*)ws;    ws += align256((size_t)HID * HID * sizeof(_Float16));

  float* out = (float*)d_out;

  const float* xcur = x0;
  int K = 128;  // IN_DIM for layer 0, then HID
  for (int l = 0; l < 3; ++l) {
    const int n_elem = N_NODES * K;

    zero_kernel<<<(n_elem + 255) / 256, 256, 0, stream>>>(agg, n_elem);

    long long tot = (long long)N_EDGES * (K / 4);
    unsigned nblk = (unsigned)((tot + 255) / 256);
    scatter_add_kernel<<<nblk, 256, 0, stream>>>(xcur, src, dst, agg, K);

    combine_cast_kernel<<<(n_elem + 255) / 256, 256, 0, stream>>>(xcur, agg, eps[l], hf, n_elem);

    transpose_cast_kernel<<<(K * HID + 255) / 256, 256, 0, stream>>>(W1[l], w1t, K);
    transpose_cast_kernel<<<(HID * HID + 255) / 256, 256, 0, stream>>>(W2[l], w2t, HID);

    dim3 grid(N_NODES / 16, 1);  // 2500 row tiles; 4 waves/block cover 256 cols
    wmma_gemm_relu<_Float16><<<grid, 128, 0, stream>>>(hf, w1t, b1[l], tf, K);
    wmma_gemm_relu<float><<<grid, 128, 0, stream>>>(tf, w2t, b2[l], out, HID);

    xcur = out;  // layer output (f32, in d_out) feeds next layer
    K = HID;
  }
}
